// TemporalGATEncoder_19121194401863
// MI455X (gfx1250) — matrix operand order repaired and checked by reading, not compiled, not run on hardware
//
#include <hip/hip_runtime.h>
#include <hip/hip_bf16.h>
#include <math.h>

#define N_NODES 20000
#define E_EDGES 320000
#define ET_EDGES (E_EDGES + N_NODES)
#define LN_EPS 1e-5f
#define NEG_SLOPE 0.2f

typedef __attribute__((ext_vector_type(16))) __bf16 v16bf;
typedef __attribute__((ext_vector_type(8)))  __bf16 v8bf;
typedef __attribute__((ext_vector_type(8)))  float  v8f;

__device__ __forceinline__ __bf16 f2bf(float f) {
  unsigned u = __builtin_bit_cast(unsigned, f);
  unsigned r = (u + 0x7FFFu + ((u >> 16) & 1u)) >> 16;   // round-to-nearest-even
  unsigned short s = (unsigned short)r;
  return __builtin_bit_cast(__bf16, s);
}

__device__ __forceinline__ float wave_sum32(float v) {
#pragma unroll
  for (int o = 16; o > 0; o >>= 1) v += __shfl_xor(v, o, 32);
  return v;
}

__device__ __forceinline__ float elu_f(float o) {
  return o > 0.f ? o : (__expf(o) - 1.f);
}

// ---------------------------------------------------------------------------
// C[M x 256] = concat_K(A0, A1)[M x K] @ B[K x 256]   (bf16 WMMA, f32 accum)
// A0/A1: bf16 row-major [M x 256]; Bt: bf16 column-major [256 x K].
// grid: (M/16, 256/128), block: 256 = 8 waves, one 16x16 tile per wave.
// ---------------------------------------------------------------------------
template <int K>
__global__ void gemm_bf16_wmma(const __bf16* __restrict__ A0,
                               const __bf16* __restrict__ A1,
                               const __bf16* __restrict__ Bt,
                               float* __restrict__ C) {
  const int wave    = threadIdx.x >> 5;
  const int lane    = threadIdx.x & 31;
  const int rowTile = blockIdx.x;
  const int colTile = blockIdx.y * 8 + wave;
  const int half    = lane >> 4;          // 0: lanes 0-15, 1: lanes 16-31
  const int l16     = lane & 15;
  const int m       = rowTile * 16 + l16;
  const int ncol    = colTile * 16 + l16;
  const int kgA     = half * 16;          // A frag K-group base (contiguous)
  const int kgB     = half * 16;          // B frag K-group base
  const __bf16* arow0 = A0 + m * 256;
  const __bf16* arow1 = A1 + m * 256;
  const __bf16* bcol  = Bt + ncol * K;

  v8f acc = {0.f, 0.f, 0.f, 0.f, 0.f, 0.f, 0.f, 0.f};
#pragma unroll
  for (int k0 = 0; k0 < K; k0 += 32) {
    const __bf16* ap = (k0 < 256) ? (arow0 + k0) : (arow1 + (k0 - 256));
    v8bf lo = *(const v8bf*)(ap + kgA);        // K = k0+kgA+0..7
    v8bf hi = *(const v8bf*)(ap + kgA + 8);    // K = k0+kgA+8..15
    v16bf a = __builtin_shufflevector(lo, hi, 0, 1, 2, 3, 4, 5, 6, 7,
                                      8, 9, 10, 11, 12, 13, 14, 15);
    v16bf b = *(const v16bf*)(bcol + k0 + kgB);
    if (k0 + 32 < K) __builtin_prefetch((const void*)(bcol + k0 + 32 + kgB), 0, 1);
    acc = __builtin_amdgcn_wmma_f32_16x16x32_bf16(false, a, false, b,
                                                  (short)0, acc, false, false);
  }
#pragma unroll
  for (int r = 0; r < 8; ++r) {           // C/D: VGPR r -> row r (+8 for half 1)
    int row = rowTile * 16 + half * 8 + r;
    C[row * 256 + ncol] = acc[r];
  }
}

// ---------------------------------------------------------------------------
__global__ void fill_f32(float* __restrict__ p, float v, int n) {
  int t = blockIdx.x * blockDim.x + threadIdx.x;
  if (t < n) p[t] = v;
}

__global__ void cvt_bf16_kernel(const float* __restrict__ in, __bf16* __restrict__ out, int n) {
  int i = blockIdx.x * blockDim.x + threadIdx.x;
  if (i < n) out[i] = f2bf(in[i]);
}

// Wt[n*K + k] = bf16(Wm[k*256 + n])   (convert + transpose to column-major)
__global__ void cvt_w_transpose(const float* __restrict__ Wm, __bf16* __restrict__ Wt, int K) {
  int t = blockIdx.x * blockDim.x + threadIdx.x;
  if (t >= 256 * K) return;
  int n = t / K, k = t - n * K;
  Wt[t] = f2bf(Wm[k * 256 + n]);
}

// M_l[f,h] = sum_c le_w[f, h*64+c] * att_edge[h,c]   (32x4 per layer)
__global__ void precompute_M(const float* __restrict__ le0, const float* __restrict__ ae0,
                             const float* __restrict__ le1, const float* __restrict__ ae1,
                             float* __restrict__ M0, float* __restrict__ M1) {
  int t = threadIdx.x;                 // 256 threads
  int which = t >> 7;
  int idx = t & 127;
  int f = idx >> 2, h = idx & 3;
  const float* le = which ? le1 : le0;
  const float* ae = which ? ae1 : ae0;
  float s = 0.f;
#pragma unroll 8
  for (int c = 0; c < 64; ++c) s += le[f * 256 + h * 64 + c] * ae[h * 64 + c];
  (which ? M1 : M0)[idx] = s;
}

// Temporal MLP + edge projection -> ef[32]; a_edge per layer; efsum for loop fill
__global__ void edge_mlp_kernel(const float* __restrict__ times,
                                const float* __restrict__ eattr,
                                const float* __restrict__ tw1, const float* __restrict__ tb1,
                                const float* __restrict__ tw2, const float* __restrict__ tb2,
                                const float* __restrict__ epw, const float* __restrict__ epb,
                                const float* __restrict__ eprw, const float* __restrict__ eprb,
                                const float* __restrict__ M0, const float* __restrict__ M1,
                                float* __restrict__ aedge0, float* __restrict__ aedge1,
                                float* __restrict__ efsum) {
  int e = blockIdx.x * blockDim.x + threadIdx.x;
  bool valid = e < E_EDGES;
  float ef[32];
  if (valid) {
    float tm = times[e];
    float t1[16];
#pragma unroll
    for (int j = 0; j < 16; ++j) { float v = tm * tw1[j] + tb1[j]; t1[j] = v > 0.f ? v : 0.f; }
    float t2[16];
#pragma unroll
    for (int j = 0; j < 16; ++j) {
      float s = tb2[j];
#pragma unroll
      for (int i = 0; i < 16; ++i) s += t1[i] * tw2[i * 16 + j];
      t2[j] = s;
    }
    float er[16];
#pragma unroll
    for (int i = 0; i < 16; ++i) er[i] = eattr[e * 16 + i];
#pragma unroll
    for (int j = 0; j < 32; ++j) {
      float s = epb[j];
#pragma unroll
      for (int i = 0; i < 16; ++i) s += t2[i] * epw[i * 32 + j];
      float a = eprb[j];
#pragma unroll
      for (int i = 0; i < 16; ++i) a += er[i] * eprw[i * 32 + j];
      ef[j] = (s > 0.f ? s : 0.f) + a;
    }
#pragma unroll
    for (int h = 0; h < 4; ++h) {
      float s0 = 0.f, s1 = 0.f;
#pragma unroll
      for (int f = 0; f < 32; ++f) { s0 += ef[f] * M0[f * 4 + h]; s1 += ef[f] * M1[f * 4 + h]; }
      aedge0[e * 4 + h] = s0;
      aedge1[e * 4 + h] = s1;
    }
  } else {
#pragma unroll
    for (int j = 0; j < 32; ++j) ef[j] = 0.f;
  }
  // wave32 reduction of ef, one atomic per component per wave
#pragma unroll
  for (int j = 0; j < 32; ++j) {
    float v = ef[j];
#pragma unroll
    for (int o = 16; o > 0; o >>= 1) v += __shfl_down(v, o, 32);
    if ((threadIdx.x & 31) == 0) atomicAdd(&efsum[j], v);
  }
}

__global__ void aloop_kernel(const float* __restrict__ efsum,
                             const float* __restrict__ M0, const float* __restrict__ M1,
                             float* __restrict__ aloop0, float* __restrict__ aloop1) {
  int h = threadIdx.x;
  if (h >= 4) return;
  float s0 = 0.f, s1 = 0.f;
  for (int f = 0; f < 32; ++f) {
    float m = efsum[f] * (1.0f / (float)E_EDGES);
    s0 += m * M0[f * 4 + h];
    s1 += m * M1[f * 4 + h];
  }
  aloop0[h] = s0;
  aloop1[h] = s1;
}

// a_src[n,h], a_dst[n,h] from h-matrix
__global__ void node_att_kernel(const float* __restrict__ Hm,
                                const float* __restrict__ att_src,
                                const float* __restrict__ att_dst,
                                float* __restrict__ asrc, float* __restrict__ adst) {
  int t = blockIdx.x * blockDim.x + threadIdx.x;
  if (t >= N_NODES * 4) return;
  int n = t >> 2, h = t & 3;
  const float* hp = Hm + n * 256 + h * 64;
  const float* as = att_src + h * 64;
  const float* ad = att_dst + h * 64;
  float s = 0.f, d = 0.f;
#pragma unroll 8
  for (int c = 0; c < 64; ++c) { float v = hp[c]; s += v * as[c]; d += v * ad[c]; }
  asrc[t] = s;
  adst[t] = d;
}

// pass A: alpha = leaky_relu(a_src[s]+a_dst[d]+a_edge), segment max (f32 atomic max)
__global__ void alpha_max_kernel(const int* __restrict__ ei,
                                 const float* __restrict__ asrc, const float* __restrict__ adst,
                                 const float* __restrict__ aedge, const float* __restrict__ aloop,
                                 float* __restrict__ alpha, float* __restrict__ amax) {
  int e = blockIdx.x * blockDim.x + threadIdx.x;
  if (e >= ET_EDGES) return;
  int s, d;
  float ae[4];
  if (e < E_EDGES) {
    s = ei[e]; d = ei[E_EDGES + e];
#pragma unroll
    for (int h = 0; h < 4; ++h) ae[h] = aedge[e * 4 + h];
  } else {
    s = d = e - E_EDGES;
#pragma unroll
    for (int h = 0; h < 4; ++h) ae[h] = aloop[h];
  }
#pragma unroll
  for (int h = 0; h < 4; ++h) {
    float a = asrc[s * 4 + h] + adst[d * 4 + h] + ae[h];
    a = a > 0.f ? a : NEG_SLOPE * a;
    alpha[e * 4 + h] = a;
    __hip_atomic_fetch_max(&amax[d * 4 + h], a, __ATOMIC_RELAXED, __HIP_MEMORY_SCOPE_AGENT);
  }
}

// pass B: ealpha = exp(alpha - amax[d]); denom[d] += ealpha
__global__ void exp_sum_kernel(const int* __restrict__ ei,
                               float* __restrict__ alpha, const float* __restrict__ amax,
                               float* __restrict__ denom) {
  int e = blockIdx.x * blockDim.x + threadIdx.x;
  if (e >= ET_EDGES) return;
  int d = (e < E_EDGES) ? ei[E_EDGES + e] : e - E_EDGES;
#pragma unroll
  for (int h = 0; h < 4; ++h) {
    float v = __expf(alpha[e * 4 + h] - amax[d * 4 + h]);
    alpha[e * 4 + h] = v;
    atomicAdd(&denom[d * 4 + h], v);
  }
}

// pass C: agg[d] += h[s] * (ealpha / (denom[d]+1e-16)); 32 lanes x 8 channels / edge
__global__ void aggregate_kernel(const int* __restrict__ ei,
                                 const float* __restrict__ Hm,
                                 const float* __restrict__ alpha,
                                 const float* __restrict__ denom,
                                 float* __restrict__ agg) {
  int tid = blockIdx.x * blockDim.x + threadIdx.x;   // ET*32 = 10,880,000 < 2^31
  int e = tid >> 5;
  int part = tid & 31;
  int c0 = part * 8;
  int h = c0 >> 6;
  int s, d;
  if (e < E_EDGES) { s = ei[e]; d = ei[E_EDGES + e]; }
  else             { s = d = e - E_EDGES; }
  float w = alpha[e * 4 + h] / (denom[d * 4 + h] + 1e-16f);
  const float* hp = Hm + s * 256 + c0;
  float* op = agg + d * 256 + c0;
#pragma unroll
  for (int j = 0; j < 8; ++j) atomicAdd(&op[j], hp[j] * w);
}

// x1 = elu(LN(agg + bias))  -> bf16, one wave32 per node
__global__ void post1_kernel(const float* __restrict__ agg, const float* __restrict__ bias,
                             const float* __restrict__ g, const float* __restrict__ b,
                             __bf16* __restrict__ x1bf) {
  int wave = threadIdx.x >> 5, lane = threadIdx.x & 31;
  int n = blockIdx.x * 8 + wave;
  float v[8];
  float s = 0.f;
#pragma unroll
  for (int j = 0; j < 8; ++j) { v[j] = agg[n * 256 + j * 32 + lane] + bias[j * 32 + lane]; s += v[j]; }
  s = wave_sum32(s);
  float mu = s * (1.f / 256.f);
  float q = 0.f;
#pragma unroll
  for (int j = 0; j < 8; ++j) { float dd = v[j] - mu; q += dd * dd; }
  q = wave_sum32(q);
  float inv = rsqrtf(q * (1.f / 256.f) + LN_EPS);
#pragma unroll
  for (int j = 0; j < 8; ++j) {
    int c = j * 32 + lane;
    float o = elu_f((v[j] - mu) * inv * g[c] + b[c]);
    x1bf[n * 256 + c] = f2bf(o);
  }
}

// x2 = elu(LN(agg + bias) + res + res_b) -> bf16
__global__ void post2_kernel(const float* __restrict__ agg, const float* __restrict__ bias,
                             const float* __restrict__ g, const float* __restrict__ b,
                             const float* __restrict__ res, const float* __restrict__ res_b,
                             __bf16* __restrict__ x2bf) {
  int wave = threadIdx.x >> 5, lane = threadIdx.x & 31;
  int n = blockIdx.x * 8 + wave;
  float v[8];
  float s = 0.f;
#pragma unroll
  for (int j = 0; j < 8; ++j) { v[j] = agg[n * 256 + j * 32 + lane] + bias[j * 32 + lane]; s += v[j]; }
  s = wave_sum32(s);
  float mu = s * (1.f / 256.f);
  float q = 0.f;
#pragma unroll
  for (int j = 0; j < 8; ++j) { float dd = v[j] - mu; q += dd * dd; }
  q = wave_sum32(q);
  float inv = rsqrtf(q * (1.f / 256.f) + LN_EPS);
#pragma unroll
  for (int j = 0; j < 8; ++j) {
    int c = j * 32 + lane;
    float o = (v[j] - mu) * inv * g[c] + b[c] + res[n * 256 + c] + res_b[c];
    x2bf[n * 256 + c] = f2bf(elu_f(o));
  }
}

__global__ void final_kernel(const float* __restrict__ t, const float* __restrict__ jkb,
                             float* __restrict__ out) {
  int i = blockIdx.x * blockDim.x + threadIdx.x;
  if (i >= N_NODES * 256) return;
  out[i] = elu_f(t[i] + jkb[i & 255]);
}

// ---------------------------------------------------------------------------
extern "C" void kernel_launch(void* const* d_in, const int* in_sizes, int n_in,
                              void* d_out, int out_size, void* d_ws, size_t ws_size,
                              hipStream_t stream) {
  const float* x       = (const float*)d_in[0];
  const int*   ei      = (const int*)  d_in[1];
  const float* times   = (const float*)d_in[2];
  const float* eattr   = (const float*)d_in[3];
  const float* tw1     = (const float*)d_in[4];
  const float* tb1     = (const float*)d_in[5];
  const float* tw2     = (const float*)d_in[6];
  const float* tb2     = (const float*)d_in[7];
  const float* epw     = (const float*)d_in[8];
  const float* epb     = (const float*)d_in[9];
  const float* eprw    = (const float*)d_in[10];
  const float* eprb    = (const float*)d_in[11];
  const float* res_w   = (const float*)d_in[12];
  const float* res_b   = (const float*)d_in[13];
  const float* jk_w    = (const float*)d_in[14];
  const float* jk_b    = (const float*)d_in[15];
  const float* W0      = (const float*)d_in[16];
  const float* att_s0  = (const float*)d_in[17];
  const float* att_d0  = (const float*)d_in[18];
  const float* le_w0   = (const float*)d_in[19];
  const float* att_e0  = (const float*)d_in[20];
  const float* bias0   = (const float*)d_in[21];
  const float* ln_g0   = (const float*)d_in[22];
  const float* ln_b0   = (const float*)d_in[23];
  const float* W1      = (const float*)d_in[24];
  const float* att_s1  = (const float*)d_in[25];
  const float* att_d1  = (const float*)d_in[26];
  const float* le_w1   = (const float*)d_in[27];
  const float* att_e1  = (const float*)d_in[28];
  const float* bias1   = (const float*)d_in[29];
  const float* ln_g1   = (const float*)d_in[30];
  const float* ln_b1   = (const float*)d_in[31];

  const long long NF = (long long)N_NODES * 256;   // 5,120,000

  // ---- f32 scratch ----
  float* Wf     = (float*)d_ws;
  float* bufA   = Wf;                        // h1 / h2 / res output
  float* bufB   = bufA + NF;                 // aggregation target / jk gemm out
  float* aedge0 = bufB + NF;                 // E*4
  float* aedge1 = aedge0 + (long long)E_EDGES * 4;
  float* alpha  = aedge1 + (long long)E_EDGES * 4;   // ET*4
  float* asrc   = alpha + (long long)ET_EDGES * 4;
  float* adst   = asrc + N_NODES * 4;
  float* amax   = adst + N_NODES * 4;
  float* denom  = amax + N_NODES * 4;
  float* efsum  = denom + N_NODES * 4;       // 32
  float* M0     = efsum + 32;                // 128
  float* M1     = M0 + 128;                  // 128
  float* aloop0 = M1 + 128;                  // 4
  float* aloop1 = aloop0 + 4;                // 4
  long long fEnd = (aloop1 + 4) - Wf;
  fEnd = (fEnd + 63) & ~63LL;                // 256B-align bf16 region

  // ---- bf16 scratch ----
  __bf16* xbf   = (__bf16*)(Wf + fEnd);
  __bf16* x1bf  = xbf + NF;
  __bf16* x2bf  = x1bf + NF;
  __bf16* W0t   = x2bf + NF;                 // [256][256]
  __bf16* W1t   = W0t + 256 * 256;
  __bf16* resWt = W1t + 256 * 256;
  __bf16* jkWt  = resWt + 256 * 256;         // [256][512]

  const int TB = 256;
  const dim3 gemmGrid(N_NODES / 16, 2);
  const int etBlocks  = (ET_EDGES + TB - 1) / TB;
  const int aggBlocks = (ET_EDGES * 32) / TB;        // exact: 42500
  const int naBlocks  = (N_NODES * 4 + TB - 1) / TB;
  const int nfBlocks  = (int)((NF + TB - 1) / TB);

  // ---- one-time conversions ----
  cvt_bf16_kernel<<<nfBlocks, TB, 0, stream>>>(x, xbf, (int)NF);
  cvt_w_transpose<<<(256 * 256) / TB, TB, 0, stream>>>(W0, W0t, 256);
  cvt_w_transpose<<<(256 * 256) / TB, TB, 0, stream>>>(W1, W1t, 256);
  cvt_w_transpose<<<(256 * 256) / TB, TB, 0, stream>>>(res_w, resWt, 256);
  cvt_w_transpose<<<(256 * 512) / TB, TB, 0, stream>>>(jk_w, jkWt, 512);

  // ---- edge features (shared by both layers) ----
  fill_f32<<<1, 32, 0, stream>>>(efsum, 0.f, 32);
  precompute_M<<<1, 256, 0, stream>>>(le_w0, att_e0, le_w1, att_e1, M0, M1);
  edge_mlp_kernel<<<E_EDGES / TB, TB, 0, stream>>>(times, eattr, tw1, tb1, tw2, tb2,
                                                   epw, epb, eprw, eprb, M0, M1,
                                                   aedge0, aedge1, efsum);
  aloop_kernel<<<1, 32, 0, stream>>>(efsum, M0, M1, aloop0, aloop1);

  // ---- layer 1 ----
  gemm_bf16_wmma<256><<<gemmGrid, TB, 0, stream>>>(xbf, xbf, W0t, bufA);
  node_att_kernel<<<naBlocks, TB, 0, stream>>>(bufA, att_s0, att_d0, asrc, adst);
  fill_f32<<<naBlocks, TB, 0, stream>>>(amax, -3.0e38f, N_NODES * 4);
  fill_f32<<<naBlocks, TB, 0, stream>>>(denom, 0.f, N_NODES * 4);
  fill_f32<<<nfBlocks, TB, 0, stream>>>(bufB, 0.f, (int)NF);
  alpha_max_kernel<<<etBlocks, TB, 0, stream>>>(ei, asrc, adst, aedge0, aloop0, alpha, amax);
  exp_sum_kernel<<<etBlocks, TB, 0, stream>>>(ei, alpha, amax, denom);
  aggregate_kernel<<<aggBlocks, TB, 0, stream>>>(ei, bufA, alpha, denom, bufB);
  post1_kernel<<<N_NODES / 8, TB, 0, stream>>>(bufB, bias0, ln_g0, ln_b0, x1bf);

  // ---- layer 2 ----
  gemm_bf16_wmma<256><<<gemmGrid, TB, 0, stream>>>(x1bf, x1bf, W1t, bufA);
  node_att_kernel<<<naBlocks, TB, 0, stream>>>(bufA, att_s1, att_d1, asrc, adst);
  fill_f32<<<naBlocks, TB, 0, stream>>>(amax, -3.0e38f, N_NODES * 4);
  fill_f32<<<naBlocks, TB, 0, stream>>>(denom, 0.f, N_NODES * 4);
  fill_f32<<<nfBlocks, TB, 0, stream>>>(bufB, 0.f, (int)NF);
  alpha_max_kernel<<<etBlocks, TB, 0, stream>>>(ei, asrc, adst, aedge1, aloop1, alpha, amax);
  exp_sum_kernel<<<etBlocks, TB, 0, stream>>>(ei, alpha, amax, denom);
  aggregate_kernel<<<aggBlocks, TB, 0, stream>>>(ei, bufA, alpha, denom, bufB);

  // residual GEMM (bufA free after aggregate)
  gemm_bf16_wmma<256><<<gemmGrid, TB, 0, stream>>>(x1bf, x1bf, resWt, bufA);
  post2_kernel<<<N_NODES / 8, TB, 0, stream>>>(bufB, bias1, ln_g1, ln_b1, bufA, res_b, x2bf);

  // ---- JK head: [x1 | x2] @ jk_w (K=512 split-A), bias + ELU ----
  gemm_bf16_wmma<512><<<gemmGrid, TB, 0, stream>>>(x1bf, x2bf, jkWt, bufB);
  final_kernel<<<nfBlocks, TB, 0, stream>>>(bufB, jk_b, (float*)d_out);
}
